// MSDGCL_80041010528268
// MI455X (gfx1250) — compile-verified, matrix operand order checked by reading, not run on hardware
//
#include <hip/hip_runtime.h>

typedef __bf16 bf16_t;
typedef __attribute__((ext_vector_type(16))) __bf16 v16bf;
typedef __attribute__((ext_vector_type(8)))  float  v8f;
typedef __attribute__((ext_vector_type(4)))  unsigned int u32x4;
typedef __attribute__((ext_vector_type(8)))  int i32x8;
typedef __attribute__((ext_vector_type(4)))  int i32x4;
typedef unsigned int u32;
typedef unsigned long long u64;

#define EPSV 1e-5f

// ---------------------------------------------------------------------------
// Block-level sum reduction (blockDim <= 256)
// ---------------------------------------------------------------------------
__device__ inline float blockReduceSum(float v) {
    __shared__ float red[256];
    __syncthreads();
    red[threadIdx.x] = v;
    __syncthreads();
    for (int s = blockDim.x >> 1; s > 0; s >>= 1) {
        if ((int)threadIdx.x < s) red[threadIdx.x] += red[threadIdx.x + s];
        __syncthreads();
    }
    return red[0];
}

// ---------------------------------------------------------------------------
// TDM: load one 16-row x 32-col bf16 tile (row stride kd elems) into LDS.
// D# built per CDNA5 ISA; tracked by TENSORcnt.
// (6-arg builtin form: g0, g1, g2, g3, g4, cpol)
// ---------------------------------------------------------------------------
__device__ inline void tdm_load_tile(const bf16_t* gsrc, u32 lds_off, int kd) {
    u64 ga = (u64)(size_t)gsrc;
    u32x4 g0;
    g0[0] = 1u;                                            // count=1, user descriptor
    g0[1] = lds_off;                                       // lds_addr (bytes)
    g0[2] = (u32)(ga & 0xffffffffu);                       // global_addr[31:0]
    g0[3] = ((u32)(ga >> 32) & 0x01ffffffu) | (2u << 30);  // addr[56:32] | type=2
    u32 td0 = (u32)kd;              // tensor_dim0 (elems)
    u32 td1 = 0x00100000u;          // tensor_dim1 (large; no OOB clip)
    u64 st0 = (u64)(u32)kd;         // tensor_dim0_stride (elems)
    i32x8 g1;
    g1[0] = (int)(1u << 16);                               // data_size=1 -> 2 bytes
    g1[1] = (int)((td0 & 0xffffu) << 16);                  // tensor_dim0[15:0]
    g1[2] = (int)((td0 >> 16) | ((td1 & 0xffffu) << 16));  // dim0[31:16] | dim1[15:0]
    g1[3] = (int)((td1 >> 16) | (32u << 16));              // dim1[31:16] | tile_dim0=32
    g1[4] = (int)16u;                                      // tile_dim1=16, tile_dim2=0
    g1[5] = (int)(u32)(st0 & 0xffffffffu);                 // dim0_stride[31:0]
    g1[6] = (int)(u32)(st0 >> 32);                         // dim0_stride[47:32]
    g1[7] = 0;                                             // dim1_stride (unused, 2D)
    i32x4 gz4 = {0, 0, 0, 0};
    i32x8 gz8 = {0, 0, 0, 0, 0, 0, 0, 0};
    __builtin_amdgcn_tensor_load_to_lds(g0, g1, gz4, gz4, gz8, 0);
}

// ---------------------------------------------------------------------------
// WMMA bf16 GEMM:  C(f32, MxN) = A(bf16, MxK, row-major) @ B^T
//   B is stored N x K row-major (transposed operand).
// Staging: A via per-lane GLOBAL_LOAD_ASYNC_TO_LDS_B128 (ASYNCcnt);
//          B via one TDM 16x32 slice per wave (TENSORcnt).
// Double-buffered LDS (2 x 16KB); one barrier per K-step; async/TDM of
// tile k+1 overlaps the 8 WMMAs of tile k.
// Block: 256 threads = 8 waves; block tile 128x128; wave tile 32x64.
// Requires M%128==0, N%128==0, K%32==0.
// ---------------------------------------------------------------------------
__global__ __launch_bounds__(256) void k_gemm_bf16(
    const bf16_t* __restrict__ A, const bf16_t* __restrict__ B,
    float* __restrict__ C, int M, int Ncols, int Kd)
{
    __shared__ __align__(16) bf16_t As[2][128][32];   // [buf][m][k]
    __shared__ __align__(16) bf16_t Bs[2][128][32];   // [buf][n][k]

    const int tid  = threadIdx.x;
    const int lane = tid & 31;
    const int wave = tid >> 5;
    const int bm = blockIdx.y * 128;
    const int bn = blockIdx.x * 128;
    const int wm = (wave >> 1) * 32;   // 0,32,64,96
    const int wn = (wave &  1) * 64;   // 0,64
    const int g   = lane >> 4;         // lane half
    const int r16 = lane & 15;

    v8f acc[2][4] = {};

    auto stage = [&](int buf, int k0) {
        // A tile: 128 rows x 64B = 512 x b128; 256 threads -> 2 each
        #pragma unroll
        for (int t = 0; t < 2; ++t) {
            int idx = t * 256 + tid;         // 0..511
            int row = idx >> 2;              // 0..127
            int seg = (idx & 3) << 3;        // bf16 offset: 0,8,16,24
            const bf16_t* ga = A + (size_t)(bm + row) * Kd + k0 + seg;
            u32 la = (u32)(size_t)&As[buf][row][seg];
            asm volatile("global_load_async_to_lds_b128 %0, %1, off"
                         :: "v"(la), "v"(ga) : "memory");
        }
        // B tile: one TDM 16x32 slice per wave
        const bf16_t* gb = B + (size_t)(bn + wave * 16) * Kd + k0;
        u32 lb = (u32)(size_t)&Bs[buf][wave * 16][0];
        tdm_load_tile(gb, lb, Kd);
    };

    stage(0, 0);
    for (int k0 = 0; k0 < Kd; k0 += 32) {
        const int buf = (k0 >> 5) & 1;
        asm volatile("s_wait_asynccnt 0" ::: "memory");
        __builtin_amdgcn_s_wait_tensorcnt(0);
        __syncthreads();
        if (k0 + 32 < Kd) stage(buf ^ 1, k0 + 32);

        // ---- fragments per CDNA5 16-bit A/B VGPR layout ----
        v16bf a0, a1, bfr[4];
        #pragma unroll
        for (int vv = 0; vv < 8; ++vv) {
            int kk = ((vv >> 2) << 4) + (g << 3) + ((vv & 3) << 1);
            a0[2*vv]   = As[buf][wm      + r16][kk];
            a0[2*vv+1] = As[buf][wm      + r16][kk + 1];
            a1[2*vv]   = As[buf][wm + 16 + r16][kk];
            a1[2*vv+1] = As[buf][wm + 16 + r16][kk + 1];
            #pragma unroll
            for (int j = 0; j < 4; ++j) {
                bfr[j][2*vv]   = Bs[buf][wn + 16*j + r16][kk];
                bfr[j][2*vv+1] = Bs[buf][wn + 16*j + r16][kk + 1];
            }
        }
        #pragma unroll
        for (int j = 0; j < 4; ++j) {
            acc[0][j] = __builtin_amdgcn_wmma_f32_16x16x32_bf16(
                false, a0, false, bfr[j], (short)0, acc[0][j], false, false);
            acc[1][j] = __builtin_amdgcn_wmma_f32_16x16x32_bf16(
                false, a1, false, bfr[j], (short)0, acc[1][j], false, false);
        }
    }

    // ---- store D per CDNA5 f32 C/D layout ----
    #pragma unroll
    for (int i = 0; i < 2; ++i) {
        #pragma unroll
        for (int j = 0; j < 4; ++j) {
            int row = bm + wm + i * 16 + g * 8;
            int col = bn + wn + j * 16 + r16;
            #pragma unroll
            for (int r = 0; r < 8; ++r)
                C[(size_t)(row + r) * Ncols + col] = acc[i][j][r];
        }
    }
}

// ---------------------------------------------------------------------------
// Elementwise / reduction helper kernels
// ---------------------------------------------------------------------------
__global__ void k_f32_to_bf16(const float* __restrict__ x, bf16_t* __restrict__ y, size_t n) {
    for (size_t i = (size_t)blockIdx.x * blockDim.x + threadIdx.x; i < n;
         i += (size_t)gridDim.x * blockDim.x)
        y[i] = (bf16_t)x[i];
}

// dst (C x R, bf16) = transpose of src (R x C, f32); R,C multiples of 32
__global__ __launch_bounds__(256) void k_convT(const float* __restrict__ src,
                                               bf16_t* __restrict__ dst, int R, int C) {
    __shared__ float tile[32][33];
    int rb = blockIdx.y * 32, cb = blockIdx.x * 32;
    int tx = threadIdx.x & 31, ty = threadIdx.x >> 5;   // 8 row-groups
    for (int rr = ty; rr < 32; rr += 8)
        tile[rr][tx] = src[(size_t)(rb + rr) * C + cb + tx];
    __syncthreads();
    for (int rr = ty; rr < 32; rr += 8)
        dst[(size_t)(cb + rr) * R + rb + tx] = (bf16_t)tile[tx][rr];
}

__global__ void k_relu_bf(const float* __restrict__ in, float* __restrict__ out,
                          bf16_t* __restrict__ outb, size_t n) {
    for (size_t i = (size_t)blockIdx.x * blockDim.x + threadIdx.x; i < n;
         i += (size_t)gridDim.x * blockDim.x) {
        float v = fmaxf(in[i], 0.f);
        if (out)  out[i]  = v;
        if (outb) outb[i] = (bf16_t)v;
    }
}

// per-column mean / var over M rows (row-major MxC)
__global__ void k_colstats(const float* __restrict__ x, float* __restrict__ mean,
                           float* __restrict__ var, int M, int C) {
    int c = blockIdx.x * blockDim.x + threadIdx.x;
    if (c >= C) return;
    float s = 0.f, s2 = 0.f;
    for (int r = 0; r < M; ++r) {
        float v = x[(size_t)r * C + c];
        s += v; s2 += v * v;
    }
    float m = s / (float)M;
    mean[c] = m;
    var[c]  = s2 / (float)M - m * m;
}

__global__ void k_bn_relu(const float* __restrict__ x, const float* __restrict__ mean,
                          const float* __restrict__ var, float* __restrict__ y,
                          bf16_t* __restrict__ yb, size_t n, int C) {
    for (size_t i = (size_t)blockIdx.x * blockDim.x + threadIdx.x; i < n;
         i += (size_t)gridDim.x * blockDim.x) {
        int c = (int)(i % (size_t)C);
        float v = (x[i] - mean[c]) * rsqrtf(var[c] + EPSV);
        v = fmaxf(v, 0.f);
        if (y)  y[i]  = v;
        if (yb) yb[i] = (bf16_t)v;
    }
}

// act: 0 = none, 1 = relu, 2 = sigmoid
__global__ void k_bias_act(const float* __restrict__ x, const float* __restrict__ bias,
                           float* __restrict__ y, bf16_t* __restrict__ yb,
                           size_t n, int C, int act) {
    for (size_t i = (size_t)blockIdx.x * blockDim.x + threadIdx.x; i < n;
         i += (size_t)gridDim.x * blockDim.x) {
        int c = (int)(i % (size_t)C);
        float v = x[i] + (bias ? bias[c] : 0.f);
        if (act == 1) v = fmaxf(v, 0.f);
        else if (act == 2) v = 1.f / (1.f + __expf(-v));
        if (y)  y[i]  = v;
        if (yb) yb[i] = (bf16_t)v;
    }
}

__global__ void k_concat3(const float* __restrict__ a, int ca,
                          const float* __restrict__ b, int cb,
                          const float* __restrict__ c, int cc,
                          float* __restrict__ out, bf16_t* __restrict__ outb, int M) {
    int C = ca + cb + cc;
    size_t total = (size_t)M * C;
    for (size_t i = (size_t)blockIdx.x * blockDim.x + threadIdx.x; i < total;
         i += (size_t)gridDim.x * blockDim.x) {
        int row = (int)(i / (size_t)C);
        int col = (int)(i % (size_t)C);
        float v;
        if (col < ca)            v = a[(size_t)row * ca + col];
        else if (col < ca + cb)  v = b[(size_t)row * cb + (col - ca)];
        else                     v = c[(size_t)row * cc + (col - ca - cb)];
        if (out)  out[i]  = v;
        if (outb) outb[i] = (bf16_t)v;
    }
}

// out[rows[e]] += vals[e] * s[cols[e]]  (one block per edge)
__global__ __launch_bounds__(256) void k_spmm(
    const int* __restrict__ rows, const int* __restrict__ cols,
    const float* __restrict__ vals, const float* __restrict__ s,
    float* __restrict__ out, int E, int C) {
    int e = blockIdx.x;
    if (e >= E) return;
    __shared__ int sr, sc; __shared__ float sv;
    if (threadIdx.x == 0) { sr = rows[e]; sc = cols[e]; sv = vals[e]; }
    __syncthreads();
    const float* src = s + (size_t)sc * C;
    float* dst = out + (size_t)sr * C;
    for (int j = threadIdx.x; j < C; j += blockDim.x)
        atomicAdd(&dst[j], sv * src[j]);
}

__global__ __launch_bounds__(256) void k_rowsq(const bf16_t* __restrict__ f,
                                               float* __restrict__ sq, int C) {
    int row = blockIdx.x;
    float s = 0.f;
    for (int j = threadIdx.x; j < C; j += blockDim.x) {
        float v = (float)f[(size_t)row * C + j];
        s += v * v;
    }
    s = blockReduceSum(s);
    if (threadIdx.x == 0) sq[row] = s;
}

// per-row top-(K+1) nearest (smallest sq[j]-2*gram[i][j]); drop self; write K ids
__global__ __launch_bounds__(128) void k_knn(const float* __restrict__ gram,
                                             const float* __restrict__ sq,
                                             int* __restrict__ nbr, int n, int kout) {
    const int KP = 11;
    int i = blockIdx.x;
    float bd[KP]; int bj[KP];
    #pragma unroll
    for (int t = 0; t < KP; ++t) { bd[t] = 3.4e38f; bj[t] = 0; }
    for (int j = threadIdx.x; j < n; j += blockDim.x) {
        float d = sq[j] - 2.f * gram[(size_t)i * n + j];
        if (d < bd[KP - 1]) {
            int t = KP - 1;
            while (t > 0 && bd[t - 1] > d) { bd[t] = bd[t - 1]; bj[t] = bj[t - 1]; --t; }
            bd[t] = d; bj[t] = j;
        }
    }
    __shared__ float cd[128 * KP];
    __shared__ int   cj[128 * KP];
    for (int t = 0; t < KP; ++t) {
        cd[threadIdx.x * KP + t] = bd[t];
        cj[threadIdx.x * KP + t] = bj[t];
    }
    __syncthreads();
    if (threadIdx.x == 0) {
        const int ncand = 128 * KP;
        int sel[KP];
        for (int t = 0; t < KP; ++t) {
            float best = 3.4e38f; int bi = 0;
            for (int c = 0; c < ncand; ++c)
                if (cd[c] < best) { best = cd[c]; bi = c; }
            sel[t] = cj[bi]; cd[bi] = 3.4e38f;
        }
        int w = 0, skipped = 0;
        for (int t = 0; t < KP && w < kout; ++t) {
            if (!skipped && sel[t] == i) { skipped = 1; continue; }
            nbr[(size_t)i * kout + (w++)] = sel[t];
        }
    }
}

__global__ void k_build_adj(const int* __restrict__ nbr, float* __restrict__ A,
                            int n, int kout) {
    int i = blockIdx.x;
    int t = threadIdx.x;
    if (t < kout) {
        int j = nbr[(size_t)i * kout + t];
        A[(size_t)i * n + j] = 1.f;
        A[(size_t)j * n + i] = 1.f;     // symmetrize (max with transpose of 0/1)
    }
}

__global__ __launch_bounds__(256) void k_degree(const float* __restrict__ A,
                                                float* __restrict__ dinv, int n) {
    int i = blockIdx.x;
    float s = 0.f;
    for (int j = threadIdx.x; j < n; j += blockDim.x) s += A[(size_t)i * n + j];
    s = blockReduceSum(s);
    if (threadIdx.x == 0) dinv[i] = (s > 0.f) ? rsqrtf(s) : 0.f;
}

__global__ void k_norm_adj(const float* __restrict__ A, bf16_t* __restrict__ Ab,
                           const float* __restrict__ dinv, int n) {
    size_t total = (size_t)n * n;
    for (size_t i = (size_t)blockIdx.x * blockDim.x + threadIdx.x; i < total;
         i += (size_t)gridDim.x * blockDim.x) {
        int r = (int)(i / (size_t)n);
        int c = (int)(i % (size_t)n);
        Ab[i] = (bf16_t)(A[i] * dinv[r] * dinv[c]);
    }
}

// r = emb + attn; row layernorm with gamma/beta; write f32 + bf16
__global__ __launch_bounds__(256) void k_attn_ln(
    const float* __restrict__ emb, const float* __restrict__ attn,
    const float* __restrict__ gg, const float* __restrict__ bb,
    float* __restrict__ out, bf16_t* __restrict__ outb, int C) {
    int row = blockIdx.x;
    const float inv = 1.f / (float)C;
    float s = 0.f;
    for (int j = threadIdx.x; j < C; j += blockDim.x)
        s += emb[(size_t)row * C + j] + attn[(size_t)row * C + j];
    float m = blockReduceSum(s) * inv;
    float s2 = 0.f;
    for (int j = threadIdx.x; j < C; j += blockDim.x) {
        float r = emb[(size_t)row * C + j] + attn[(size_t)row * C + j] - m;
        s2 += r * r;
    }
    float rstd = rsqrtf(blockReduceSum(s2) * inv + EPSV);
    for (int j = threadIdx.x; j < C; j += blockDim.x) {
        float r = emb[(size_t)row * C + j] + attn[(size_t)row * C + j];
        float o = gg[j] * (r - m) * rstd + bb[j];
        out[(size_t)row * C + j]  = o;
        outb[(size_t)row * C + j] = (bf16_t)o;
    }
}

__global__ __launch_bounds__(256) void k_mse(const float* __restrict__ a,
                                             const float* __restrict__ b,
                                             float* __restrict__ out, size_t n, float scale) {
    float s = 0.f;
    for (size_t i = (size_t)blockIdx.x * blockDim.x + threadIdx.x; i < n;
         i += (size_t)gridDim.x * blockDim.x) {
        float d = a[i] - b[i];
        s += d * d;
    }
    s = blockReduceSum(s);
    if (threadIdx.x == 0) atomicAdd(out, s * scale);
}

// ---------------------------------------------------------------------------
// Host orchestration
// ---------------------------------------------------------------------------
extern "C" void kernel_launch(void* const* d_in, const int* in_sizes, int n_in,
                              void* d_out, int out_size, void* d_ws, size_t ws_size,
                              hipStream_t stream) {
    (void)in_sizes; (void)n_in; (void)out_size; (void)ws_size;
    constexpr int N = 4096, F = 2048, H = 256, K = 10, E = 131072;
    constexpr int H2 = 512, H3 = 768, H4 = 1024, H7 = 1792;

    const float* x        = (const float*)d_in[0];
    const int*   adj_rows = (const int*)d_in[1];
    const int*   adj_cols = (const int*)d_in[2];
    const float* adj_vals = (const float*)d_in[3];
    const int*   adjd_rows= (const int*)d_in[4];
    const int*   adjd_cols= (const int*)d_in[5];
    const float* adjd_vals= (const float*)d_in[6];
    const float* W1 = (const float*)d_in[7];
    const float* W2 = (const float*)d_in[9];
    const float* W3 = (const float*)d_in[11];
    const float* D1 = (const float*)d_in[13];
    const float* d1 = (const float*)d_in[14];
    const float* D2 = (const float*)d_in[15];
    const float* d2 = (const float*)d_in[16];
    const float* D3 = (const float*)d_in[17];
    const float* d3 = (const float*)d_in[18];
    const float* G1 = (const float*)d_in[19];
    const float* G2 = (const float*)d_in[20];
    const float* G3 = (const float*)d_in[21];
    const float* Wv = (const float*)d_in[22];
    const float* bv = (const float*)d_in[23];
    const float* Wo = (const float*)d_in[24];
    const float* bo = (const float*)d_in[25];
    const float* ln_g = (const float*)d_in[26];
    const float* ln_b = (const float*)d_in[27];
    const float* P1 = (const float*)d_in[28];
    const float* P2 = (const float*)d_in[30];
    const float* p2b = (const float*)d_in[31];
    // b1,b2,b3,p1b feed _bn and cancel exactly (column shift removed by mean)

    float* out_pred = (float*)d_out;
    float* out_mse  = out_pred + (size_t)N * N;
    float* out_emb  = out_mse + 1;

    char* wsp = (char*)d_ws;
    auto alloc  = [&](size_t bytes) -> void* {
        void* p = (void*)wsp;
        wsp += (bytes + 255) & ~((size_t)255);
        return p;
    };
    auto allocF = [&](size_t n) { return (float*)alloc(n * sizeof(float)); };
    auto allocB = [&](size_t n) { return (bf16_t*)alloc(n * sizeof(bf16_t)); };
    auto allocI = [&](size_t n) { return (int*)alloc(n * sizeof(int)); };

    auto ew = [](size_t n) -> unsigned {
        size_t b = (n + 255) / 256;
        if (b > 4096) b = 4096;
        if (b < 1) b = 1;
        return (unsigned)b;
    };
    auto conv = [&](const float* src, bf16_t* dst, size_t n) {
        k_f32_to_bf16<<<ew(n), 256, 0, stream>>>(src, dst, n);
    };
    // dst(bf16, C x R) = src(f32, R x C)^T
    auto convT = [&](const float* src, bf16_t* dst, int R, int C) {
        dim3 grid((unsigned)(C / 32), (unsigned)(R / 32));
        k_convT<<<grid, 256, 0, stream>>>(src, dst, R, C);
    };
    auto gemm = [&](const bf16_t* A, const bf16_t* B, float* C, int M, int Nc, int Kd) {
        dim3 grid((unsigned)(Nc / 128), (unsigned)(M / 128));
        k_gemm_bf16<<<grid, 256, 0, stream>>>(A, B, C, M, Nc, Kd);
    };

    // ---- workspace carve ----
    bf16_t* xb  = allocB((size_t)N * F);
    bf16_t* w1b = allocB((size_t)F * H4);     // stored H4 x F
    bf16_t* w2b = allocB((size_t)H4 * H2);    // H2 x H4
    bf16_t* w3b = allocB((size_t)H2 * H);     // H  x H2
    bf16_t* d1b = allocB((size_t)H * H2);     // H2 x H
    bf16_t* d2b = allocB((size_t)H2 * H4);    // H4 x H2
    bf16_t* d3b = allocB((size_t)H4 * F);     // F  x H4
    bf16_t* g1b = allocB((size_t)H7 * H4);    // H4 x H7
    bf16_t* g2b = allocB((size_t)H4 * H2);    // H2 x H4
    bf16_t* g3b = allocB((size_t)H2 * H);     // H  x H2
    bf16_t* wvb = allocB((size_t)H3 * H3);    // H3 x H3
    bf16_t* wob = allocB((size_t)H3 * H3);    // H3 x H3
    bf16_t* p1w = allocB((size_t)H3 * N);     // N  x H3
    bf16_t* p2w = allocB((size_t)N * N);      // N  x N

    float*  z1f = allocF((size_t)N * H4);  bf16_t* z1b = allocB((size_t)N * H4);
    float*  z2f = allocF((size_t)N * H2);  bf16_t* z2b = allocB((size_t)N * H2);
    float*  z3f = allocF((size_t)N * H);   bf16_t* z3b = allocB((size_t)N * H);
    float*  meanb = allocF(N);
    float*  varb  = allocF(N);

    float*  t1  = allocF((size_t)N * H2);  bf16_t* t1b = allocB((size_t)N * H2);
    float*  t2  = allocF((size_t)N * H4);  bf16_t* t2b = allocB((size_t)N * H4);
    float*  xd  = allocF((size_t)N * F);

    bf16_t* f_db = allocB((size_t)N * H7);
    float*  sqb  = allocF(N);
    int*    nbr  = allocI((size_t)N * K);
    float*  dinv = allocF(N);

    float*  BIG  = allocF((size_t)N * N);    // gram -> A_knn -> P1-out -> P2-out
    bf16_t* Abf  = allocB((size_t)N * N);

    float*  sf  = allocF((size_t)N * H4);
    bf16_t* sbT = allocB((size_t)N * H4);    // H4 x N (transposed for dense B)
    float*  p   = allocF((size_t)N * H4);  bf16_t* pb  = allocB((size_t)N * H4);
    float*  q   = allocF((size_t)N * H4);  bf16_t* qbT = allocB((size_t)N * H4);
    float*  h3  = allocF((size_t)N * H);
    float*  hk3 = allocF((size_t)N * H);
    float*  hd3 = allocF((size_t)N * H);

    float*  ecat = allocF((size_t)N * H3); bf16_t* ecatb = allocB((size_t)N * H3);
    float*  vf   = allocF((size_t)N * H3); bf16_t* vfb   = allocB((size_t)N * H3);
    float*  af_  = allocF((size_t)N * H3);
    bf16_t* elnb = allocB((size_t)N * H3);
    bf16_t* qpb  = allocB((size_t)N * N);

    // ---- bf16 mirrors (weights transposed to N x K) ----
    conv(x,  xb,  (size_t)N * F);
    convT(W1, w1b, F,  H4);   convT(W2, w2b, H4, H2);
    convT(W3, w3b, H2, H);    convT(D1, d1b, H,  H2);
    convT(D2, d2b, H2, H4);   convT(D3, d3b, H4, F);
    convT(G1, g1b, H7, H4);   convT(G2, g2b, H4, H2);
    convT(G3, g3b, H2, H);    convT(Wv, wvb, H3, H3);
    convT(Wo, wob, H3, H3);   convT(P1, p1w, H3, N);
    convT(P2, p2w, N,  N);

    auto bnrelu = [&](float* buf, bf16_t* bufb, int Cc) {
        k_colstats<<<(Cc + 255) / 256, 256, 0, stream>>>(buf, meanb, varb, N, Cc);
        k_bn_relu<<<ew((size_t)N * Cc), 256, 0, stream>>>(buf, meanb, varb, buf, bufb,
                                                          (size_t)N * Cc, Cc);
    };

    // ---- encoder (bias cancels inside _bn) ----
    gemm(xb,  w1b, z1f, N, H4, F);  bnrelu(z1f, z1b, H4);
    gemm(z1b, w2b, z2f, N, H2, H4); bnrelu(z2f, z2b, H2);
    gemm(z2b, w3b, z3f, N, H,  H2); bnrelu(z3f, z3b, H);

    // ---- decoder + MSE ----
    gemm(z3b, d1b, t1, N, H2, H);
    k_bias_act<<<ew((size_t)N * H2), 256, 0, stream>>>(t1, d1, t1, t1b, (size_t)N * H2, H2, 1);
    gemm(t1b, d2b, t2, N, H4, H2);
    k_bias_act<<<ew((size_t)N * H4), 256, 0, stream>>>(t2, d2, t2, t2b, (size_t)N * H4, H4, 1);
    gemm(t2b, d3b, xd, N, F, H4);
    k_bias_act<<<ew((size_t)N * F), 256, 0, stream>>>(xd, d3, xd, (bf16_t*)nullptr,
                                                      (size_t)N * F, F, 1);
    (void)hipMemsetAsync(out_mse, 0, sizeof(float), stream);
    k_mse<<<1024, 256, 0, stream>>>(xd, x, out_mse, (size_t)N * F, 1.f / ((float)N * F));

    // ---- f_d, knn adjacency ----
    k_concat3<<<ew((size_t)N * H7), 256, 0, stream>>>(z1f, H4, z2f, H2, z3f, H,
                                                      (float*)nullptr, f_db, N);
    k_rowsq<<<N, 256, 0, stream>>>(f_db, sqb, H7);
    gemm(f_db, f_db, BIG, N, N, H7);                       // gram = f_d @ f_d^T
    k_knn<<<N, 128, 0, stream>>>(BIG, sqb, nbr, N, K);
    (void)hipMemsetAsync(BIG, 0, (size_t)N * N * sizeof(float), stream);
    k_build_adj<<<N, 32, 0, stream>>>(nbr, BIG, N, K);
    k_degree<<<N, 256, 0, stream>>>(BIG, dinv, N);
    k_norm_adj<<<4096, 256, 0, stream>>>(BIG, Abf, dinv, N);

    // ---- s = f_d @ G1 ----
    gemm(f_db, g1b, sf, N, H4, H7);

    // ---- sparse branches (adj, adjd) ----
    auto branch_sp = [&](const int* rws, const int* cls, const float* vls, float* hout) {
        (void)hipMemsetAsync(p, 0, (size_t)N * H4 * sizeof(float), stream);
        k_spmm<<<E, 256, 0, stream>>>(rws, cls, vls, sf, p, E, H4);
        k_relu_bf<<<ew((size_t)N * H4), 256, 0, stream>>>(p, p, pb, (size_t)N * H4);
        gemm(pb, g2b, q, N, H2, H4);
        (void)hipMemsetAsync(p, 0, (size_t)N * H2 * sizeof(float), stream);
        k_spmm<<<E, 256, 0, stream>>>(rws, cls, vls, q, p, E, H2);
        k_relu_bf<<<ew((size_t)N * H2), 256, 0, stream>>>(p, p, pb, (size_t)N * H2);
        gemm(pb, g3b, q, N, H, H2);
        (void)hipMemsetAsync(p, 0, (size_t)N * H * sizeof(float), stream);
        k_spmm<<<E, 256, 0, stream>>>(rws, cls, vls, q, p, E, H);
        k_relu_bf<<<ew((size_t)N * H), 256, 0, stream>>>(p, hout, (bf16_t*)nullptr, (size_t)N * H);
    };
    branch_sp(adj_rows, adj_cols, adj_vals, h3);

    // ---- knn dense branch (WMMA through A_knn; dense B transposed) ----
    convT(sf, sbT, N, H4);                        // s^T : H4 x N
    gemm(Abf, sbT, p, N, H4, N);
    k_relu_bf<<<ew((size_t)N * H4), 256, 0, stream>>>(p, p, pb, (size_t)N * H4);
    gemm(pb, g2b, q, N, H2, H4);
    convT(q, qbT, N, H2);                         // (hk@G2)^T : H2 x N
    gemm(Abf, qbT, p, N, H2, N);
    k_relu_bf<<<ew((size_t)N * H2), 256, 0, stream>>>(p, p, pb, (size_t)N * H2);
    gemm(pb, g3b, q, N, H, H2);
    convT(q, qbT, N, H);                          // (hk@G3)^T : H x N
    gemm(Abf, qbT, p, N, H, N);
    k_relu_bf<<<ew((size_t)N * H), 256, 0, stream>>>(p, hk3, (bf16_t*)nullptr, (size_t)N * H);

    branch_sp(adjd_rows, adjd_cols, adjd_vals, hd3);

    // ---- attention + layernorm ----
    k_concat3<<<ew((size_t)N * H3), 256, 0, stream>>>(h3, H, hk3, H, hd3, H, ecat, ecatb, N);
    gemm(ecatb, wvb, vf, N, H3, H3);
    k_bias_act<<<ew((size_t)N * H3), 256, 0, stream>>>(vf, bv, vf, vfb, (size_t)N * H3, H3, 0);
    gemm(vfb, wob, af_, N, H3, H3);
    k_bias_act<<<ew((size_t)N * H3), 256, 0, stream>>>(af_, bo, af_, (bf16_t*)nullptr,
                                                       (size_t)N * H3, H3, 0);
    k_attn_ln<<<N, 256, 0, stream>>>(ecat, af_, ln_g, ln_b, out_emb, elnb, H3);

    // ---- prediction head (p1b cancels in _bn) ----
    gemm(elnb, p1w, BIG, N, N, H3);
    k_colstats<<<(N + 255) / 256, 256, 0, stream>>>(BIG, meanb, varb, N, N);
    k_bn_relu<<<4096, 256, 0, stream>>>(BIG, meanb, varb, (float*)nullptr, qpb,
                                        (size_t)N * N, N);
    gemm(qpb, p2w, BIG, N, N, N);
    k_bias_act<<<4096, 256, 0, stream>>>(BIG, p2b, out_pred, (bf16_t*)nullptr,
                                         (size_t)N * N, N, 2);
}
